// NeRFRenderer_50586124812839
// MI455X (gfx1250) — matrix operand order repaired and checked by reading, not compiled, not run on hardware
//
#include <hip/hip_runtime.h>
#include <hip/hip_bf16.h>

// NeRF renderer for MI455X (gfx1250, wave32, WMMA).
// Pipeline (all on `stream`):
//   K1 nerf_composite<64,true>  : generate stratified z_coarse (hash RNG),
//                                 fused MLP (layer1 VALU f32 -> layer2 WMMA f16),
//                                 alpha composite -> rgb_c, depth_c, w_coarse(ws), z_coarse(ws)
//   K2 fine_sample              : per-ray CDF + inverse-CDF sampling + 128-elem bitonic sort -> z_all(ws)
//   K3 nerf_composite<128,false>: same fused MLP/composite over z_all -> rgb_f, depth_f, w_fine
//
// Output layout (floats): rgb_c[B*3] | depth_c[B] | rgb_f[B*3] | depth_f[B] | w_fine[B*128]
// Workspace (floats):     z_coarse[B*64] | w_coarse[B*64] | z_all[B*128]   (= 32 MB)

#define NB        32768
#define NCOARSE   64
#define NFINE     64
#define HIDDEN    64

typedef __attribute__((ext_vector_type(16))) _Float16 v16h;
typedef __attribute__((ext_vector_type(8)))  float    v8f;

// Deterministic counter-based RNG (lowbias32 hash) -> [0,1)
__device__ __forceinline__ float u01(unsigned int x) {
  x ^= x >> 17; x *= 0xed5ad4bbu;
  x ^= x >> 11; x *= 0xac4c1b51u;
  x ^= x >> 15; x *= 0x31848babu;
  x ^= x >> 14;
  return (float)(x >> 8) * (1.0f / 16777216.0f);
}

// ---------------------------------------------------------------------------
// Fused MLP + alpha-composite kernel.
// One wave (32 lanes) == one ray. 256 threads/block = 8 waves = 8 rays.
// Samples processed in tiles of 16 (WMMA M dimension).
//   layer1: h = relu(p @ W1 + b1)  -- K=3, done in f32 VALU, each lane produces
//           exactly the halves its WMMA A-fragment needs (ISA 7.12.2 layout).
//   layer2: out = h @ W2pad + b2   -- two v_wmma_f32_16x16x32_f16 (K=64), f32 acc.
//   composite: exclusive cumprod scan over 16 lanes + cross-tile T carry.
// ---------------------------------------------------------------------------
template <int K, bool GEN_Z>
__global__ __launch_bounds__(256)
void nerf_composite(const float* __restrict__ rays,
                    const float* __restrict__ W1, const float* __restrict__ b1,
                    const float* __restrict__ W2, const float* __restrict__ b2,
                    float* __restrict__ zbuf_g,      // GEN_Z: write; else: read
                    float* __restrict__ w_out,       // [B,K] weights
                    float* __restrict__ rgb_out,     // [B,3]
                    float* __restrict__ depth_out)   // [B]
{
  const int tid  = threadIdx.x;
  const int wave = tid >> 5;
  const int lane = tid & 31;
  const int g    = lane >> 4;   // half-wave group (0/1)
  const int m    = lane & 15;   // row (sample within tile) / output column
  const int ray  = blockIdx.x * 8 + wave;   // grid sized exactly: NB/8 blocks

  // --- weights staged in LDS (heavy reuse: 32 hidden units/lane/tile) ---
  __shared__ float sW[192 + 64 + 4];         // W1(3x64) | b1(64) | b2(4)
  __shared__ float sOut[8 * 64];             // per-wave 16x4 WMMA result tile
  for (int i = tid; i < 260; i += 256)
    sW[i] = (i < 192) ? W1[i] : ((i < 256) ? b1[i - 192] : b2[i - 256]);
  __syncthreads();
  const float* sW1 = sW;
  const float* sb1 = sW + 192;
  const float* sb2 = sW + 256;

  // --- per-ray constants (uniform per wave -> scalarized loads) ---
  const long rb = (long)ray * 8;
  const float ox = rays[rb + 0], oy = rays[rb + 1], oz = rays[rb + 2];
  const float dx = rays[rb + 3], dy = rays[rb + 4], dz = rays[rb + 5];
  const float zn = rays[rb + 6], zf = rays[rb + 7];

  // --- B fragments: W2 (64x4) zero-padded to 64x16, split into two K=32 tiles.
  //     ISA layout: lane = column N=m (group g); half e = row K = 8g+e (e<8), 16+8g+(e-8) (e>=8).
  v16h bLo, bHi;
#pragma unroll
  for (int e = 0; e < 16; ++e) {
    const int kk = ((e < 8) ? e : (e + 8)) + 8 * g;
    bLo[e] = (_Float16)((m < 4) ? W2[kk * 4 + m]        : 0.0f);
    bHi[e] = (_Float16)((m < 4) ? W2[(kk + 32) * 4 + m] : 0.0f);
  }
  const float c0 = (m < 4) ? sb2[m] : 0.0f;  // bias preloaded into accumulator

  float rAcc = 0.f, gAcc = 0.f, bAcc = 0.f, dAcc = 0.f;
  float Tc = 1.0f;  // transmittance carry across tiles

  constexpr int T = K / 16;
#pragma unroll 1
  for (int t = 0; t < T; ++t) {
    const int s = t * 16 + m;  // global sample index owned by this lane-row

    // --- z for this sample and the next (delta) ---
    float zs, zs1;
    if (GEN_Z) {  // stratified coarse sampling: z_step = (i + u)/N
      const float u = u01((unsigned)(ray * K + s) * 0x9e3779b9u + 0x1234567u);
      zs = zn + (zf - zn) * ((float)s + u) * (1.0f / (float)K);
      if (s + 1 < K) {
        const float u2 = u01((unsigned)(ray * K + s + 1) * 0x9e3779b9u + 0x1234567u);
        zs1 = zn + (zf - zn) * ((float)(s + 1) + u2) * (1.0f / (float)K);
      } else zs1 = zf;
    } else {
      zs  = zbuf_g[(long)ray * K + s];
      zs1 = (s + 1 < K) ? zbuf_g[(long)ray * K + s + 1] : zf;
    }

    // --- layer 1 (K=3, f32 VALU) straight into the A-fragment layout ---
    const float px = ox + zs * dx, py = oy + zs * dy, pz = oz + zs * dz;
    v16h aLo, aHi;
#pragma unroll
    for (int e = 0; e < 16; ++e) {
      const int j = ((e < 8) ? e : (e + 8)) + 8 * g;      // hidden col for K-chunk 0..31
      float h0 = fmaf(pz, sW1[128 + j], fmaf(py, sW1[64 + j], fmaf(px, sW1[j], sb1[j])));
      aLo[e] = (_Float16)fmaxf(h0, 0.0f);
      const int j2 = j + 32;                               // hidden col 32..63
      float h1 = fmaf(pz, sW1[128 + j2], fmaf(py, sW1[64 + j2], fmaf(px, sW1[j2], sb1[j2])));
      aHi[e] = (_Float16)fmaxf(h1, 0.0f);
    }

    // --- layer 2: two chained WMMAs, f32 accumulate (EXEC all-1s here) ---
    v8f acc = {c0, c0, c0, c0, c0, c0, c0, c0};
    acc = __builtin_amdgcn_wmma_f32_16x16x32_f16(false, aLo, false, bLo,
                                                 (short)0, acc, false, false);
    acc = __builtin_amdgcn_wmma_f32_16x16x32_f16(false, aHi, false, bHi,
                                                 (short)0, acc, false, false);

    // --- scatter 16x4 result tile to per-wave LDS (D layout: row r+8g, col m) ---
    if (m < 4) {
#pragma unroll
      for (int r = 0; r < 8; ++r)
        sOut[wave * 64 + (r + 8 * g) * 4 + m] = acc[r];
    }
    __syncthreads();  // uniform: exact grid sizing, no early-exit

    // --- composite: lane-row m owns sample s ---
    const float cr = sOut[wave * 64 + m * 4 + 0];
    const float cg = sOut[wave * 64 + m * 4 + 1];
    const float cb = sOut[wave * 64 + m * 4 + 2];
    const float sg = sOut[wave * 64 + m * 4 + 3];

    const float delta = zs1 - zs;
    const float alpha = 1.0f - __expf(-delta * fmaxf(sg, 0.0f));
    const float tin   = 1.0f - alpha + 1e-10f;

    // inclusive product scan over 16 lanes (segments of 16 within wave32)
    float p = tin;
#pragma unroll
    for (int d = 1; d < 16; d <<= 1) {
      const float o = __shfl_up(p, d, 16);
      if (m >= d) p *= o;
    }
    float excl = __shfl_up(p, 1, 16);
    if (m == 0) excl = 1.0f;

    const float w = alpha * Tc * excl;
    rAcc += w * cr; gAcc += w * cg; bAcc += w * cb; dAcc += w * zs;
    if (lane < 16) {
      w_out[(long)ray * K + s] = w;
      if (GEN_Z) zbuf_g[(long)ray * K + s] = zs;
    }
    Tc *= __shfl(p, 15, 16);  // total tile transmittance -> carry
  }

  // --- reduce the 16 per-sample partials and write ray outputs ---
#pragma unroll
  for (int d = 8; d >= 1; d >>= 1) {
    rAcc += __shfl_down(rAcc, d, 16);
    gAcc += __shfl_down(gAcc, d, 16);
    bAcc += __shfl_down(bAcc, d, 16);
    dAcc += __shfl_down(dAcc, d, 16);
  }
  if (lane == 0) {
    rgb_out[(long)ray * 3 + 0] = rAcc;
    rgb_out[(long)ray * 3 + 1] = gAcc;
    rgb_out[(long)ray * 3 + 2] = bAcc;
    depth_out[ray] = dAcc;
  }
}

// ---------------------------------------------------------------------------
// Fine sampling: one 128-thread block per ray.
//   CDF over coarse weights -> inverse-CDF sample 64 fine z -> merge with
//   64 coarse z -> 128-element bitonic sort in LDS -> z_all.
// ---------------------------------------------------------------------------
__global__ __launch_bounds__(128)
void fine_sample(const float* __restrict__ rays,
                 const float* __restrict__ w_coarse,   // [B,64]
                 const float* __restrict__ z_coarse,   // [B,64]
                 float* __restrict__ z_all)            // [B,128]
{
  const int tid = threadIdx.x;
  const int ray = blockIdx.x;

  __shared__ float s_pdf[64];
  __shared__ float s_cdf[65];
  __shared__ float s_z[128];

  const float zn = rays[(long)ray * 8 + 6];
  const float zf = rays[(long)ray * 8 + 7];

  if (tid < 64) {
    s_pdf[tid] = w_coarse[(long)ray * 64 + tid] + 1e-5f;
    s_z[tid]   = z_coarse[(long)ray * 64 + tid];
  }
  __syncthreads();

  // serial 64-elem prefix sum + normalize (negligible vs. rest)
  if (tid == 0) {
    float c = 0.0f;
    s_cdf[0] = 0.0f;
    for (int i = 0; i < 64; ++i) { c += s_pdf[i]; s_cdf[i + 1] = c; }
    const float inv = 1.0f / c;
    for (int i = 1; i <= 64; ++i) s_cdf[i] *= inv;
  }
  __syncthreads();

  if (tid < 64) {
    const unsigned base = (unsigned)(ray * 64 + tid);
    const float u  = u01(base * 0x85ebca6bu + 0xdeadbeefu);
    const float u2 = u01(base * 0xc2b2ae35u + 0xcafef00du);
    // searchsorted(cdf, u, 'right'): first index with cdf[idx] > u
    int lo = 0, hi = 65;
    while (lo < hi) {
      const int mid = (lo + hi) >> 1;
      if (s_cdf[mid] > u) hi = mid; else lo = mid + 1;
    }
    const float indf = fmaxf((float)lo - 1.0f, 0.0f);
    const float zstep = (indf + u2) * (1.0f / (float)NCOARSE);
    s_z[64 + tid] = zn * (1.0f - zstep) + zf * zstep;
  }
  __syncthreads();

  // bitonic sort of 128 values; one element per thread
  for (int k = 2; k <= 128; k <<= 1) {
    for (int j = k >> 1; j > 0; j >>= 1) {
      const int ixj = tid ^ j;
      if (ixj > tid) {
        const float a = s_z[tid], b = s_z[ixj];
        const bool up = ((tid & k) == 0);
        if (up ? (a > b) : (a < b)) { s_z[tid] = b; s_z[ixj] = a; }
      }
      __syncthreads();
    }
  }

  z_all[(long)ray * 128 + tid] = s_z[tid];
}

// ---------------------------------------------------------------------------
extern "C" void kernel_launch(void* const* d_in, const int* in_sizes, int n_in,
                              void* d_out, int out_size, void* d_ws, size_t ws_size,
                              hipStream_t stream) {
  (void)in_sizes; (void)n_in; (void)out_size; (void)ws_size;

  const float* rays = (const float*)d_in[0];   // [B,8]
  const float* W1   = (const float*)d_in[1];   // [3,64]
  const float* b1   = (const float*)d_in[2];   // [64]
  const float* W2   = (const float*)d_in[3];   // [64,4]
  const float* b2   = (const float*)d_in[4];   // [4]

  float* out = (float*)d_out;
  float* rgb_c   = out;                          // B*3
  float* depth_c = out + (long)NB * 3;           // B
  float* rgb_f   = out + (long)NB * 4;           // B*3
  float* depth_f = out + (long)NB * 7;           // B
  float* w_fine  = out + (long)NB * 8;           // B*128

  float* ws       = (float*)d_ws;
  float* z_coarse = ws;                          // B*64
  float* w_coarse = ws + (long)NB * 64;          // B*64
  float* z_all    = ws + (long)NB * 128;         // B*128

  // One wave per ray: NB/8 blocks of 256 threads (exact -> uniform barriers).
  nerf_composite<NCOARSE, true><<<NB / 8, 256, 0, stream>>>(
      rays, W1, b1, W2, b2, z_coarse, w_coarse, rgb_c, depth_c);

  fine_sample<<<NB, 128, 0, stream>>>(rays, w_coarse, z_coarse, z_all);

  nerf_composite<NCOARSE + NFINE, false><<<NB / 8, 256, 0, stream>>>(
      rays, W1, b1, W2, b2, z_all, w_fine, rgb_f, depth_f);
}
